// UnquantizedSparseMoELayer_20220706029971
// MI455X (gfx1250) — matrix operand order repaired and checked by reading, not compiled, not run on hardware
//
#include <hip/hip_runtime.h>

#define T_TOK 32768
#define D_DIM 1024
#define E_NUM 8
#define DFF_  2048
#define MTILE 16
#define MAX_MT (T_TOK / MTILE)
#define XSTRIDE 1032 /* 1024 + two 4-elem pads (TDM pad: 2 dwords per 256 dwords) */

typedef __attribute__((ext_vector_type(16))) __bf16 v16bf;
typedef __attribute__((ext_vector_type(8)))  float  v8f;

union FragA { v16bf v; uint2 d[4]; };  // 8B-aligned chunks (x tile has 8B mid-row pad)
union FragB { v16bf v; uint4 q[2]; };  // 16B-aligned chunks

#if __has_builtin(__builtin_amdgcn_tensor_load_to_lds) && __has_builtin(__builtin_amdgcn_s_wait_tensorcnt)
#define USE_TDM 1
typedef unsigned int tdm_u32x4 __attribute__((ext_vector_type(4)));
typedef int          tdm_i32x8 __attribute__((ext_vector_type(8)));
typedef int          tdm_i32x4 __attribute__((ext_vector_type(4)));
#else
#define USE_TDM 0
#endif

__device__ __forceinline__ int xcol(int k) { return k + ((k >> 9) << 2); } // skip 4-elem pad at k=512

__device__ __forceinline__ __bf16 f2bf(float f) {
  union { float f; unsigned u; } v; v.f = f;
  unsigned r = v.u + 0x7FFFu + ((v.u >> 16) & 1u); // round-to-nearest-even
  unsigned short h = (unsigned short)(r >> 16);
  __bf16 b; __builtin_memcpy(&b, &h, sizeof(b));
  return b;
}

// ---------------- one-time fp32 -> bf16 repack (bandwidth-bound)
__global__ void cvt_f32_to_bf16(const float* __restrict__ src, __bf16* __restrict__ dst, int n8) {
  int i = blockIdx.x * blockDim.x + threadIdx.x;
  if (i >= n8) return;
  const float4* s = (const float4*)src + 2 * (size_t)i;
  float4 f0 = s[0], f1 = s[1];
  union { __bf16 b[8]; uint4 q; } o;
  o.b[0] = f2bf(f0.x); o.b[1] = f2bf(f0.y); o.b[2] = f2bf(f0.z); o.b[3] = f2bf(f0.w);
  o.b[4] = f2bf(f1.x); o.b[5] = f2bf(f1.y); o.b[6] = f2bf(f1.z); o.b[7] = f2bf(f1.w);
  *((uint4*)dst + i) = o.q;
}

// ---------------- routing: softmax over 8 logits, top-2, compact per-expert lists
__global__ void moe_route(const float* __restrict__ gate,
                          int* __restrict__ pos,
                          int* __restrict__ slot_tok,
                          float* __restrict__ slot_w) {
  int t = blockIdx.x * blockDim.x + threadIdx.x;
  if (t >= T_TOK) return;
  float l[E_NUM];
  float mx = -1e30f;
#pragma unroll
  for (int e = 0; e < E_NUM; ++e) { l[e] = gate[t * E_NUM + e]; mx = fmaxf(mx, l[e]); }
#pragma unroll
  for (int e = 0; e < E_NUM; ++e) l[e] = __expf(l[e] - mx);
  int e0 = 0; float p0 = l[0];
#pragma unroll
  for (int e = 1; e < E_NUM; ++e) if (l[e] > p0) { p0 = l[e]; e0 = e; }
  int e1 = (e0 == 0) ? 1 : 0; float p1 = l[e1];
#pragma unroll
  for (int e = 0; e < E_NUM; ++e) if (e != e0 && l[e] > p1) { p1 = l[e]; e1 = e; }
  float inv = 1.f / (p0 + p1);
  int s0 = atomicAdd(&pos[e0], 1);
  slot_tok[e0 * T_TOK + s0] = t;  slot_w[e0 * T_TOK + s0] = p0 * inv;
  int s1 = atomicAdd(&pos[e1], 1);
  slot_tok[e1 * T_TOK + s1] = t;  slot_w[e1 * T_TOK + s1] = p1 * inv;
}

// ---------------- fused expert FFN, all-bf16 WMMA, TDM-gathered x tile
struct Smem {
  __bf16 x_lds[MTILE][XSTRIDE]; // 16 gathered token rows, TDM pad layout (33KB)
  __bf16 h_lds[MTILE][136];     // 16x128 h tile, padded stride (4.3KB)
  int    s_tok[MTILE];
  float  s_w[MTILE];
};

__global__ __launch_bounds__(256, 2)
void moe_ffn(const __bf16* __restrict__ xb,
             const __bf16* __restrict__ w13b,
             const __bf16* __restrict__ w2b,
             const int* __restrict__ pos,
             const int* __restrict__ slot_tok,
             const float* __restrict__ slot_w,
             float* __restrict__ out) {
  const int e  = blockIdx.y;
  const int mt = blockIdx.x;
  const int count = pos[e];
  if (mt * MTILE >= count) return; // uniform early-exit before any barrier

  __shared__ Smem sm;
  const int tid = threadIdx.x;
  if (tid < MTILE) {
    int idx = mt * MTILE + tid;
    bool valid = idx < count;
    sm.s_tok[tid] = valid ? slot_tok[e * T_TOK + idx] : 0;
    sm.s_w[tid]   = valid ? slot_w[e * T_TOK + idx]   : 0.f;
  }
  __syncthreads();

  const int wave = tid >> 5;
  const int lane = tid & 31;
  const int half = lane >> 4;
  const int mr   = lane & 15;

#if USE_TDM
  // One TDM gather op: 16 gathered token rows (2KB each) -> LDS, with LDS padding
  if (wave == 0) {
    unsigned long long ga = (unsigned long long)(const void*)xb;
    tdm_u32x4 g0;
    g0[0] = 0x80000001u;  // count=1, gather_mode=1, 16-bit indices
    g0[1] = 0u;           // lds_addr: x_lds is at offset 0 of the static LDS block
    g0[2] = (unsigned)ga;
    g0[3] = (unsigned)((ga >> 32) & 0x01FFFFFFu) | (2u << 30); // addr[56:32] | type=2
    tdm_i32x8 g1;
    g1[0] = (int)((1u << 16)      // data_size = 2B
                | (1u << 20)      // pad_enable
                | (7u << 22)      // pad_interval: every 256 dwords
                | (1u << 25));    // pad_amount: 2 dwords
    g1[1] = (int)((D_DIM & 0xFFFFu) << 16);                  // tensor_dim0 lo
    g1[2] = (int)((D_DIM >> 16) | ((T_TOK & 0xFFFF) << 16)); // dim0 hi | tensor_dim1 lo
    g1[3] = (int)(((unsigned)T_TOK >> 16) | ((unsigned)D_DIM << 16)); // dim1 hi | tile_dim0
    g1[4] = MTILE;   // tile_dim1 = #gather indices (16)
    g1[5] = D_DIM;   // tensor_dim0_stride (elements)
    g1[6] = 0;
    g1[7] = 0;
    tdm_i32x4 g2, g3; // 16-bit row indices = gathered token ids
#pragma unroll
    for (int p = 0; p < 4; ++p) {
      g2[p] = (sm.s_tok[2 * p]     & 0xFFFF) | (sm.s_tok[2 * p + 1] << 16);
      g3[p] = (sm.s_tok[8 + 2 * p] & 0xFFFF) | (sm.s_tok[9 + 2 * p] << 16);
    }
    tdm_i32x8 g4 = {0, 0, 0, 0, 0, 0, 0, 0}; // extra group (zero-filled, 6-arg form)
    __builtin_amdgcn_tensor_load_to_lds(g0, g1, g2, g3, g4, 0);
    __builtin_amdgcn_s_wait_tensorcnt(0);
  }
#else
  // fallback: cooperative synchronous staging with identical padded layout
  for (int i = tid; i < MTILE * (D_DIM / 4); i += 256) {
    int row = i >> 8;
    int k = (i & 255) * 4;
    *(uint2*)&sm.x_lds[row][xcol(k)] =
        *(const uint2*)(xb + (size_t)sm.s_tok[row] * D_DIM + k);
  }
#endif
  __syncthreads();

  const __bf16* xr   = &sm.x_lds[mr][0];
  const __bf16* w13e = w13b + (size_t)e * 2 * DFF_ * D_DIM;
  const __bf16* w2e  = w2b  + (size_t)e * D_DIM * DFF_;

  int   tokm[8]; float wm[8];
#pragma unroll
  for (int r = 0; r < 8; ++r) { tokm[r] = sm.s_tok[half * 8 + r]; wm[r] = sm.s_w[half * 8 + r]; }

  v8f acc[8];
#pragma unroll
  for (int j = 0; j < 8; ++j) acc[j] = (v8f){};

  for (int fc = 0; fc < DFF_ / 128; ++fc) {
    const int fbase = fc * 128;
    const int fw    = fbase + wave * 16;

    v8f g = (v8f){}, u = (v8f){};
    const __bf16* bgrow = w13e + (size_t)(fw + mr) * D_DIM;
    const __bf16* burow = bgrow + (size_t)DFF_ * D_DIM;
    __builtin_prefetch(w2e + (size_t)(wave * 128 + mr) * DFF_ + fbase, 0, 0);

    for (int k0 = 0; k0 < D_DIM; k0 += 32) {
      // A 16x32: lane m=mr, K = half*16 + 0..15
      FragA a;
      int c0 = xcol(k0 + half * 16);
      int c1 = xcol(k0 + half * 16 + 8);
      a.d[0] = *(const uint2*)(xr + c0);
      a.d[1] = *(const uint2*)(xr + c0 + 4);
      a.d[2] = *(const uint2*)(xr + c1);
      a.d[3] = *(const uint2*)(xr + c1 + 4);
      // B 32x16: lane n=mr, K = half*16 + 0..15 (direct bf16 b128 loads)
      FragB bg, bu;
      const __bf16* bp = bgrow + k0 + half * 16;
      bg.q[0] = *(const uint4*)bp;
      bg.q[1] = *(const uint4*)(bp + 8);
      const __bf16* up = burow + k0 + half * 16;
      bu.q[0] = *(const uint4*)up;
      bu.q[1] = *(const uint4*)(up + 8);
      g = __builtin_amdgcn_wmma_f32_16x16x32_bf16(false, a.v, false, bg.v, (short)0, g, false, false);
      u = __builtin_amdgcn_wmma_f32_16x16x32_bf16(false, a.v, false, bu.v, (short)0, u, false, false);
    }

    // h = silu(g)*u -> LDS (fast v_rcp instead of IEEE divide)
#pragma unroll
    for (int r = 0; r < 8; ++r) {
      float gv = g[r];
      float den = 1.f + __expf(-gv);
#if __has_builtin(__builtin_amdgcn_rcpf)
      float hv = gv * __builtin_amdgcn_rcpf(den) * u[r];
#else
      float hv = gv / den * u[r];
#endif
      sm.h_lds[half * 8 + r][wave * 16 + mr] = f2bf(hv);
    }
    __syncthreads();

    // out_tile(16x1024) += h(16x128) @ w2[:, fbase:fbase+128]^T
#pragma unroll
    for (int j = 0; j < 8; ++j) {
      const __bf16* b2row = w2e + (size_t)(wave * 128 + j * 16 + mr) * DFF_ + fbase;
#pragma unroll
      for (int k2 = 0; k2 < 128; k2 += 32) {
        FragB a2, b2;
        a2.q[0] = *(const uint4*)&sm.h_lds[mr][k2 + half * 16];
        a2.q[1] = *(const uint4*)&sm.h_lds[mr][k2 + half * 16 + 8];
        const __bf16* b2p = b2row + k2 + half * 16;
        b2.q[0] = *(const uint4*)b2p;
        b2.q[1] = *(const uint4*)(b2p + 8);
        acc[j] = __builtin_amdgcn_wmma_f32_16x16x32_bf16(false, a2.v, false, b2.v, (short)0, acc[j], false, false);
      }
    }
    __syncthreads();
  }

  // epilogue: out[token, d] += routing_weight * acc (two experts/token overlap)
#pragma unroll
  for (int j = 0; j < 8; ++j) {
    int d = wave * 128 + j * 16 + mr;
#pragma unroll
    for (int r = 0; r < 8; ++r) {
      float v = acc[j][r] * wm[r];
      atomicAdd(out + (size_t)tokm[r] * D_DIM + d, v);
    }
  }
}

extern "C" void kernel_launch(void* const* d_in, const int* in_sizes, int n_in,
                              void* d_out, int out_size, void* d_ws, size_t ws_size,
                              hipStream_t stream) {
  const float* x    = (const float*)d_in[0];
  const float* gate = (const float*)d_in[1];
  const float* w13  = (const float*)d_in[2];
  const float* w2   = (const float*)d_in[3];
  float* out = (float*)d_out;

  // workspace layout (~162 MiB): routing tables + bf16 mirrors of x/w13/w2
  char*  ws = (char*)d_ws;
  size_t o = 0;
  int*    pos      = (int*)(ws + o);    o += 256;
  int*    slot_tok = (int*)(ws + o);    o += (size_t)E_NUM * T_TOK * 4;
  float*  slot_w   = (float*)(ws + o);  o += (size_t)E_NUM * T_TOK * 4;
  __bf16* xb       = (__bf16*)(ws + o); o += (size_t)T_TOK * D_DIM * 2;
  __bf16* w13b     = (__bf16*)(ws + o); o += (size_t)E_NUM * 2 * DFF_ * D_DIM * 2;
  __bf16* w2b      = (__bf16*)(ws + o); o += (size_t)E_NUM * D_DIM * DFF_ * 2;

  (void)hipMemsetAsync(pos, 0, sizeof(int) * E_NUM, stream);
  (void)hipMemsetAsync(out, 0, sizeof(float) * (size_t)out_size, stream);

  int n8x  = T_TOK * D_DIM / 8;
  int n8w1 = E_NUM * 2 * DFF_ * D_DIM / 8;
  int n8w2 = E_NUM * D_DIM * DFF_ / 8;
  cvt_f32_to_bf16<<<n8x / 256, 256, 0, stream>>>(x, xb, n8x);
  cvt_f32_to_bf16<<<n8w1 / 256, 256, 0, stream>>>(w13, w13b, n8w1);
  cvt_f32_to_bf16<<<n8w2 / 256, 256, 0, stream>>>(w2, w2b, n8w2);

  moe_route<<<T_TOK / 256, 256, 0, stream>>>(gate, pos, slot_tok, slot_w);

  dim3 grid(MAX_MT, E_NUM);
  moe_ffn<<<grid, 256, 0, stream>>>(xb, w13b, w2b, pos, slot_tok, slot_w, out);
}
